// UpdateX_12094627906027
// MI455X (gfx1250) — matrix operand order repaired and checked by reading, not compiled, not run on hardware
//
#include <hip/hip_runtime.h>
#include <hip/hip_bf16.h>
#include <stdint.h>

// Shapes from the reference
#define LSZ   256   // L
#define RSZ   128   // R (rows, also the K of stage A)
#define INDIM 128
#define DMSZ  32    // dim_msa
#define DIMK  128   // dim (output channels)

typedef __attribute__((ext_vector_type(16))) __bf16 v16bf;
typedef __attribute__((ext_vector_type(8)))  float  v8f;
typedef __attribute__((ext_vector_type(4)))  unsigned int u32x4;
typedef __attribute__((ext_vector_type(8)))  int          i32x8;
typedef __attribute__((ext_vector_type(4)))  int          i32x4;

union FragU { uint4 q[2]; v16bf v; };

__device__ __forceinline__ unsigned short f2bf(float f) {
  unsigned u = __float_as_uint(f);
  u += 0x7FFFu + ((u >> 16) & 1u);          // round-to-nearest-even
  return (unsigned short)(u >> 16);
}

// 16-bit A-matrix fragment (16x32, M x K):
// lanes 0-15: row m, K = 0..15; lanes 16-31: row m, K = 16..31 (contiguous 16)
__device__ __forceinline__ v16bf load_a_frag(const unsigned short* base, int stride,
                                             int row, int k0, int lane) {
  const unsigned short* p = base + row * stride + k0 + ((lane >> 4) << 4);
  FragU u;
  u.q[0] = *reinterpret_cast<const uint4*>(p);       // K run of 8
  u.q[1] = *reinterpret_cast<const uint4*>(p + 8);   // K run of 8, +8
  return u.v;
}

// 16-bit B-matrix fragment (32x16, K x N):
// lanes 0-15: col n, K = 0..15; lanes 16-31: col n, K = 16..31 (contiguous 16)
__device__ __forceinline__ v16bf load_b_frag(const unsigned short* base, int stride,
                                             int col, int k0, int lane) {
  const unsigned short* p = base + col * stride + k0 + ((lane >> 4) << 4);
  FragU u;
  u.q[0] = *reinterpret_cast<const uint4*>(p);
  u.q[1] = *reinterpret_cast<const uint4*>(p + 8);
  return u.v;
}

__device__ __forceinline__ v8f wmma_bf16(v16bf a, v16bf b, v8f c) {
  return __builtin_amdgcn_wmma_f32_16x16x32_bf16(
      /*neg_a=*/false, a, /*neg_b=*/false, b,
      /*c_mod=*/(short)0, c, /*reuse_a=*/false, /*reuse_b=*/false);
}

// TDM: DMA a (rows x 128)-element bf16 tile (row stride 1024 elem) from global to LDS.
// D# per CDNA5 ISA: group0 {count, lds_addr, global_addr, type=2},
// group1 {data_size=2B, tensor_dim0=1024, tensor_dim1=128, tile_dim0=128, tile_dim1=rows,
//         tensor_dim0_stride=1024}; groups 2/3 zero (2-D tile).
__device__ __forceinline__ void tdm_load_w2_chunk(const unsigned short* gsrc,
                                                  unsigned lds_byte_addr, int rows) {
  unsigned long long ga = (unsigned long long)(uintptr_t)gsrc;
  u32x4 g0;
  g0[0] = 1u;                                                    // count=1 (valid), user mode
  g0[1] = lds_byte_addr;                                         // lds_addr (bytes)
  g0[2] = (unsigned)(ga & 0xFFFFFFFFull);                        // global_addr[31:0]
  g0[3] = (unsigned)((ga >> 32) & 0x01FFFFFFull) | (2u << 30);   // global_addr[56:32] | type=2
  i32x8 g1;
  g1[0] = (1 << 16);        // workgroup_mask=0, data_size=1 (2 bytes)
  g1[1] = (1024 << 16);     // tensor_dim0[15:0] (bits 63:48)
  g1[2] = (128  << 16);     // tensor_dim1[15:0] (bits 95:80)
  g1[3] = (128  << 16);     // tile_dim0 = 128   (bits 127:112)
  g1[4] = rows;             // tile_dim1         (bits 143:128)
  g1[5] = 1024;             // tensor_dim0_stride[31:0]
  g1[6] = 0;
  g1[7] = 0;
  i32x4 gz4 = {0, 0, 0, 0};
  i32x8 gz8 = {0, 0, 0, 0, 0, 0, 0, 0};
  // 6-arg form (clang-23 / therock headers): (g0, g1, g2, g3, g4, cpol)
  __builtin_amdgcn_tensor_load_to_lds(g0, g1, gz4, gz4, gz8, 0);
}

// --------------------------------------------------------------------------
// K1: mp[r,i,d] = W1 @ m[r,i,:] + b1  ->  mpT[i][d][r] (bf16)
// --------------------------------------------------------------------------
__global__ __launch_bounds__(128)
void proj_down1_kernel(const float* __restrict__ m, const float* __restrict__ W1,
                       const float* __restrict__ b1, unsigned short* __restrict__ mpT) {
  __shared__ float sm[INDIM];
  const int bid = blockIdx.x;            // bid = r*LSZ + i
  const int r   = bid >> 8;
  const int i   = bid & (LSZ - 1);
  const int tid = threadIdx.x;
  sm[tid] = m[(size_t)bid * INDIM + tid];
  __syncthreads();
  if (tid < DMSZ) {
    float acc = b1[tid];
    for (int e = 0; e < INDIM; ++e)
      acc = fmaf(sm[e], W1[tid * INDIM + e], acc);
    mpT[((size_t)i * DMSZ + tid) * RSZ + r] = f2bf(acc);
  }
}

// --------------------------------------------------------------------------
// K2: W2 (f32, [DIMK][DMSZ*DMSZ]) -> bf16 copy in workspace
// --------------------------------------------------------------------------
__global__ __launch_bounds__(256)
void w2_to_bf16_kernel(const float* __restrict__ W2, unsigned short* __restrict__ W2b) {
  int t = blockIdx.x * blockDim.x + threadIdx.x;
  if (t < DIMK * DMSZ * DMSZ) W2b[t] = f2bf(W2[t]);
}

// --------------------------------------------------------------------------
// K3: fused  O_ij = A_i^T A_j  (stage A)  ->  out = x + b2 + W2 @ vec(O)/128
// One wave per (i, 16-wide j strip); 2 waves / block; 128 KB LDS:
//   Vsh:  2 x (16 x 1024) bf16  (per-wave V operand for stage B)
//   Wst:  2 x (128 x 128) bf16  (double-buffered W2 chunk, filled by TDM)
// Each wave TDM-loads its 64-row half of each W2 chunk; barrier publishes both.
// --------------------------------------------------------------------------
__global__ __launch_bounds__(64)
void fused_outer_proj_kernel(const float* __restrict__ x, const float* __restrict__ b2,
                             const unsigned short* __restrict__ mpT,
                             const unsigned short* __restrict__ W2b,
                             float* __restrict__ out) {
  __shared__ unsigned short Vsh[2][16 * 1024];
  __shared__ unsigned short Wst[2][DIMK * 128];

  const int wave = threadIdx.x >> 5;
  const int lane = threadIdx.x & 31;
  const int lm   = lane & 15;
  const int hi   = lane >> 4;

  const int strip = blockIdx.x * 2 + wave;       // 4096 strips total
  const int i  = strip >> 4;
  const int j0 = (strip & 15) << 4;

  unsigned short* V = Vsh[wave];
  const unsigned short* Ai = mpT + (size_t)i * (DMSZ * RSZ);
  const int r0 = wave * 64;                      // this wave's W2-row half

  // Kick off chunk 0 DMA (overlaps the whole stage-A GEMM sequence)
  tdm_load_w2_chunk(W2b + (size_t)r0 * 1024,
                    (unsigned)(uintptr_t)&Wst[0][r0 * 128], 64);

  // Preload A_i fragments (reused for all 16 j): M=d(32), K=r(128)
  v16bf afrag[4][2];
#pragma unroll
  for (int kk = 0; kk < 4; ++kk)
#pragma unroll
    for (int dt = 0; dt < 2; ++dt)
      afrag[kk][dt] = load_a_frag(Ai, RSZ, dt * 16 + lm, kk * 32, lane);

  // ---------------- Stage A: O_ij = A_i^T A_j for 16 j's ----------------
  for (int jj = 0; jj < 16; ++jj) {
    const unsigned short* Aj = mpT + (size_t)(j0 + jj) * (DMSZ * RSZ);
    v8f acc[2][2] = {};
#pragma unroll
    for (int kk = 0; kk < 4; ++kk) {
      const int k0 = kk * 32;
      v16bf b0 = load_b_frag(Aj, RSZ, lm,      k0, lane);   // c tile 0
      v16bf b1 = load_b_frag(Aj, RSZ, 16 + lm, k0, lane);   // c tile 1
      acc[0][0] = wmma_bf16(afrag[kk][0], b0, acc[0][0]);
      acc[0][1] = wmma_bf16(afrag[kk][0], b1, acc[0][1]);
      acc[1][0] = wmma_bf16(afrag[kk][1], b0, acc[1][0]);
      acc[1][1] = wmma_bf16(afrag[kk][1], b1, acc[1][1]);
    }
    // Scatter O[d,c]/128 -> V[jj][c*32 + d] as bf16 (16B packed stores)
    const float s = 1.0f / 128.0f;
#pragma unroll
    for (int dt = 0; dt < 2; ++dt)
#pragma unroll
      for (int ct = 0; ct < 2; ++ct) {
        const int c     = ct * 16 + lm;
        const int dbase = dt * 16 + hi * 8;
        unsigned short h[8];
#pragma unroll
        for (int g = 0; g < 8; ++g) h[g] = f2bf(acc[dt][ct][g] * s);
        uint4 pk;
        pk.x = (unsigned)h[0] | ((unsigned)h[1] << 16);
        pk.y = (unsigned)h[2] | ((unsigned)h[3] << 16);
        pk.z = (unsigned)h[4] | ((unsigned)h[5] << 16);
        pk.w = (unsigned)h[6] | ((unsigned)h[7] << 16);
        *reinterpret_cast<uint4*>(&V[jj * 1024 + c * 32 + dbase]) = pk;
      }
  }

  // -------- Stage B: out_strip = V (16x1024) x W2^T (1024x128), W2 via TDM --------
  v8f accB[8] = {};
  for (int ch = 0; ch < 8; ++ch) {
    __builtin_amdgcn_s_wait_tensorcnt(0);   // my half of chunk ch landed in LDS
    __syncthreads();                        // both halves of chunk ch published
    if (ch + 1 < 8)                         // buf[(ch+1)&1] was consumed before this
      tdm_load_w2_chunk(W2b + (size_t)r0 * 1024 + (ch + 1) * 128,
                        (unsigned)(uintptr_t)&Wst[(ch + 1) & 1][r0 * 128], 64);
    const unsigned short* Wc = Wst[ch & 1];
#pragma unroll
    for (int cl = 0; cl < 128; cl += 32) {
      v16bf va = load_a_frag(V, 1024, lm, ch * 128 + cl, lane);   // M = j rows (LDS)
#pragma unroll
      for (int nt = 0; nt < 8; ++nt) {
        v16bf wb = load_b_frag(Wc, 128, nt * 16 + lm, cl, lane);  // N = k cols (LDS)
        accB[nt] = wmma_bf16(va, wb, accB[nt]);
      }
    }
  }

  // Epilogue: out[i, j0+M, k] = x + b2[k] + acc
#pragma unroll
  for (int nt = 0; nt < 8; ++nt) {
    const int k  = nt * 16 + lm;
    const float bk = b2[k];
#pragma unroll
    for (int g = 0; g < 8; ++g) {
      const int jrow = j0 + hi * 8 + g;
      const size_t oidx = ((size_t)i * LSZ + jrow) * DIMK + k;
      out[oidx] = x[oidx] + accB[nt][g] + bk;
    }
  }
}

// --------------------------------------------------------------------------
extern "C" void kernel_launch(void* const* d_in, const int* in_sizes, int n_in,
                              void* d_out, int out_size, void* d_ws, size_t ws_size,
                              hipStream_t stream) {
  const float* x  = (const float*)d_in[0];
  const float* m  = (const float*)d_in[1];
  const float* W1 = (const float*)d_in[2];
  const float* b1 = (const float*)d_in[3];
  const float* W2 = (const float*)d_in[4];
  const float* b2 = (const float*)d_in[5];
  float* out = (float*)d_out;

  // Workspace: mpT (256*32*128 bf16 = 2 MB), W2 bf16 (128*1024 bf16 = 256 KB)
  unsigned short* mpT = (unsigned short*)d_ws;
  unsigned short* W2b = mpT + (size_t)LSZ * DMSZ * RSZ;

  proj_down1_kernel<<<RSZ * LSZ, 128, 0, stream>>>(m, W1, b1, mpT);
  w2_to_bf16_kernel<<<(DIMK * DMSZ * DMSZ + 255) / 256, 256, 0, stream>>>(W2, W2b);
  fused_outer_proj_kernel<<<(LSZ * 16) / 2, 64, 0, stream>>>(x, b2, mpT, W2b, out);
}
